// ColumnSelfAttention_56392920597075
// MI455X (gfx1250) — matrix operand order repaired and checked by reading, not compiled, not run on hardware
//
#include <hip/hip_runtime.h>
#include <math.h>

#define DEV __device__ __forceinline__

// ---------------------------------------------------------------------------
// Problem constants (ColumnSelfAttention: R=128, C=512, B=1, E=768, H=12, D=64)
// ---------------------------------------------------------------------------
#define R_DIM 128
#define C_DIM 512
#define E_DIM 768
#define H_DIM 12
#define D_HEAD 64
#define T_TOK (R_DIM * C_DIM)          // 65536 tokens
#define NCHUNK 32                      // split-K chunks over columns for scores
#define CPC (C_DIM / NCHUNK)           // 16 columns per chunk

typedef __attribute__((ext_vector_type(16))) __bf16 v16bf;
typedef __attribute__((ext_vector_type(8)))  float  v8f;
typedef unsigned int u32x4 __attribute__((ext_vector_type(4)));
typedef int          i32x4 __attribute__((ext_vector_type(4)));
typedef int          i32x8 __attribute__((ext_vector_type(8)));

union Frag16 { v16bf v; uint4 u4[2]; unsigned short h[16]; };
static_assert(sizeof(Frag16) == 32, "frag must be 8 VGPRs");

// round-to-nearest-even f32 -> bf16
DEV unsigned short f2bf(float f) {
  unsigned int u = __float_as_uint(f);
  u += 0x7FFFu + ((u >> 16) & 1u);
  return (unsigned short)(u >> 16);
}

DEV uint4 pack8_bf16(const float* v) {
  union { uint4 u; unsigned short h[8]; } pk;
#pragma unroll
  for (int j = 0; j < 8; ++j) pk.h[j] = f2bf(v[j]);
  return pk.u;
}

// A-operand fragment, 16x32 bf16 (M x K). ISA layout:
//   lane m      (0..15): halves[0..7]=A[m][0..7],  halves[8..15]=A[m][16..23]
//   lane m+16        : halves[0..7]=A[m][8..15], halves[8..15]=A[m][24..31]
// LDS tile is row-major [m][k], stride in halves.
DEV v16bf ldsA(const unsigned short* base, int stride, int lane) {
  int m  = lane & 15;
  int kh = lane >> 4;                 // 0 or 1
  const unsigned short* p = base + m * stride + kh * 8;
  Frag16 f;
  f.u4[0] = *(const uint4*)(p);
  f.u4[1] = *(const uint4*)(p + 16);
  return f.v;
}

// B-operand fragment, 32x16 bf16 (K x N). ISA layout (dense analog of sparse-B):
//   lane n      (0..15): halves[0..7]=B[0..7][n],  halves[8..15]=B[16..23][n]
//   lane n+16        : halves[0..7]=B[8..15][n], halves[8..15]=B[24..31][n]
// Staged transposed in LDS as rows [n][k] so the lane reads contiguous k.
DEV v16bf ldsB(const unsigned short* base, int stride, int lane) {
  int n  = lane & 15;
  int kh = lane >> 4;
  const unsigned short* p = base + n * stride + kh * 8;
  Frag16 f;
  f.u4[0] = *(const uint4*)(p);
  f.u4[1] = *(const uint4*)(p + 16);
  return f.v;
}

DEV v8f wmma_bf16(v16bf a, v16bf b, v8f c) {
  return __builtin_amdgcn_wmma_f32_16x16x32_bf16(false, a, false, b, (short)0, c,
                                                 false, false);
}

// ---------------------------------------------------------------------------
// Tensor Data Mover: async-load a 2D bf16 tile [tile_h rows x 32 elems] into
// LDS at byte offset lds_addr, with hardware LDS padding of 4 DWORDs after
// every 16 DWORDs -> padded row stride of 40 halves (80 B), matching ldsA/ldsB.
// D# built per CDNA5 ISA §8: group0 {count=1, lds_addr, global 57b, type=2},
// group1 {data_size=2B, pad_enable, pad_interval=16DW, pad_amount=4DW,
//         tensor_dim0/1 = tile dims (no OOB), tile_dim0=32, tile_dim1=tile_h,
//         tensor_dim0_stride = row pitch in elements (48-bit)}.
// ---------------------------------------------------------------------------
DEV void tdm_load_2d_bf16(unsigned lds_addr, const void* gptr, unsigned tile_h,
                          unsigned long long stride_elems) {
  unsigned long long ga = (unsigned long long)gptr;
  u32x4 g0;
  g0[0] = 1u;                                       // count=1, user descriptor
  g0[1] = lds_addr;                                 // LDS byte address
  g0[2] = (unsigned)ga;                             // global_addr[31:0]
  g0[3] = (unsigned)((ga >> 32) & 0x01FFFFFFull) |  // global_addr[56:32]
          0x80000000u;                              // type=2 ("image")
  i32x8 g1;
  g1[0] = (int)((1u << 16) |                        // data_size = 2 bytes
                (1u << 20) |                        // pad_enable
                (3u << 22) |                        // pad_interval: 16 DWORDs
                (3u << 25));                        // pad_amount:   4 DWORDs
  g1[1] = (int)(32u << 16);                         // tensor_dim0 = 32
  g1[2] = (int)(tile_h << 16);                      // tensor_dim1 = tile_h
  g1[3] = (int)(32u << 16);                         // tile_dim0 = 32
  g1[4] = (int)tile_h;                              // tile_dim1 = tile_h
  g1[5] = (int)(unsigned)stride_elems;              // dim0_stride[31:0]
  g1[6] = (int)(unsigned)(stride_elems >> 32);      // dim0_stride[47:32]
  g1[7] = 0;
  i32x4 z4 = {0, 0, 0, 0};
#if defined(__clang_major__) && __clang_major__ >= 23
  i32x8 z8 = {0, 0, 0, 0, 0, 0, 0, 0};
  __builtin_amdgcn_tensor_load_to_lds(g0, g1, z4, z4, z8, 0);
#else
  __builtin_amdgcn_tensor_load_to_lds(g0, g1, z4, z4, 0);
#endif
}

// ---------------------------------------------------------------------------
// K0: f32 -> bf16 conversion (vectorized 8 elems/thread)
// ---------------------------------------------------------------------------
__global__ void cvt_f32_bf16(const float* __restrict__ src,
                             unsigned short* __restrict__ dst, int n8) {
  int idx = blockIdx.x * blockDim.x + threadIdx.x;
  if (idx >= n8) return;
  const float4* s = (const float4*)src;
  float4 a = s[(size_t)idx * 2];
  float4 b = s[(size_t)idx * 2 + 1];
  float v[8] = {a.x, a.y, a.z, a.w, b.x, b.y, b.z, b.w};
  ((uint4*)dst)[idx] = pack8_bf16(v);
}

// ---------------------------------------------------------------------------
// K1/K4: D[feat][tok] = sum_e W[feat][e] * in[tok][e]  (i.e. in @ W^T)
// Block tile 128 feat x 128 tok, BLOCK_K=32, 8 waves (2x4), wave = 64x32 =
// 4x2 WMMA tiles. TDM double-buffered staging: wave 0 issues two tensor loads
// for the next k-slice while everyone computes on the current buffer; one
// s_wait_tensorcnt + one barrier per k-step. k-loop unrolled by 2 so the
// buffer select is a compile-time constant (no per-iteration offset loads).
// LDS (dynamic, 40960 B): Wt buf0 @0, buf1 @10240; Xt buf0 @20480, buf1 @30720.
// ---------------------------------------------------------------------------
template <bool BF16OUT>
__global__ __launch_bounds__(256) void gemm_proj(
    const unsigned short* __restrict__ in, const unsigned short* __restrict__ W,
    const float* __restrict__ bias, void* __restrict__ out, float scale) {
  extern __shared__ unsigned short smem[];
  const int tok0  = blockIdx.x * 128;
  const int feat0 = blockIdx.y * 128;

  const int tid = threadIdx.x;
  const int lane = tid & 31;
  const int w = tid >> 5;
  const int wm = w >> 2;  // feature 64-block: 0..1
  const int wn = w & 3;   // token 32-block:   0..3

  v8f acc[4][2];
#pragma unroll
  for (int r = 0; r < 4; ++r)
#pragma unroll
    for (int s = 0; s < 2; ++s)
#pragma unroll
      for (int j = 0; j < 8; ++j) acc[r][s][j] = 0.0f;

  if (w == 0) {
    tdm_load_2d_bf16(0u,      &W[(size_t)feat0 * E_DIM], 128, E_DIM);
    tdm_load_2d_bf16(20480u,  &in[(size_t)tok0 * E_DIM], 128, E_DIM);
  }
  __builtin_amdgcn_s_wait_tensorcnt(0);
  __syncthreads();

  auto kstep = [&](const unsigned cur, int k0_next, bool prefetch) {
    if (prefetch && w == 0) {
      tdm_load_2d_bf16((cur ^ 1u) * 10240u,
                       &W[(size_t)feat0 * E_DIM + k0_next], 128, E_DIM);
      tdm_load_2d_bf16(20480u + (cur ^ 1u) * 10240u,
                       &in[(size_t)tok0 * E_DIM + k0_next], 128, E_DIM);
    }
    const unsigned short* Wt = smem + cur * 5120u;
    const unsigned short* Xt = smem + 10240u + cur * 5120u;

    v16bf af[4], bfr[2];
#pragma unroll
    for (int r = 0; r < 4; ++r) af[r] = ldsA(&Wt[(wm * 64 + r * 16) * 40], 40, lane);
#pragma unroll
    for (int s = 0; s < 2; ++s) bfr[s] = ldsB(&Xt[(wn * 32 + s * 16) * 40], 40, lane);
#pragma unroll
    for (int r = 0; r < 4; ++r)
#pragma unroll
      for (int s = 0; s < 2; ++s) acc[r][s] = wmma_bf16(af[r], bfr[s], acc[r][s]);

    __builtin_amdgcn_s_wait_tensorcnt(0);  // wave0: next-buffer DMA complete
    __syncthreads();
  };

  for (int t = 0; t < 12; ++t) {          // 24 k-steps, unrolled x2
    kstep(0u, (2 * t + 1) * 32, true);
    kstep(1u, (2 * t + 2) * 32, t < 11);
  }

  const int nloc = lane & 15;
  const int hi = lane >> 4;
#pragma unroll
  for (int r = 0; r < 4; ++r) {
    int m0 = feat0 + wm * 64 + r * 16 + hi * 8;  // 8 contiguous features
    float bb[8];
    *(float4*)&bb[0] = *(const float4*)&bias[m0];
    *(float4*)&bb[4] = *(const float4*)&bias[m0 + 4];
#pragma unroll
    for (int s = 0; s < 2; ++s) {
      size_t n = (size_t)tok0 + wn * 32 + s * 16 + nloc;
      float vals[8];
#pragma unroll
      for (int j = 0; j < 8; ++j) vals[j] = (acc[r][s][j] + bb[j]) * scale;
      if (BF16OUT) {
        *(uint4*)&((unsigned short*)out)[n * E_DIM + m0] = pack8_bf16(vals);
      } else {
        float* op = &((float*)out)[n * E_DIM + m0];
        *(float4*)op       = make_float4(vals[0], vals[1], vals[2], vals[3]);
        *(float4*)(op + 4) = make_float4(vals[4], vals[5], vals[6], vals[7]);
      }
    }
  }
}

// ---------------------------------------------------------------------------
// K2: partial attention scores. Grid (NCHUNK, H). Each WG computes the full
// 128x128 S[i][j] for its head over 16 columns (K = 16*64 = 1024, 32 k-steps).
// TDM staging with the strided token pitch (512*768 elems); double buffered,
// unrolled x2 like gemm_proj.
// A = K-rows (m=j), B = Q^T (n=i) so per-lane stores are contiguous in j.
// ---------------------------------------------------------------------------
__global__ __launch_bounds__(256) void scores_partial(
    const unsigned short* __restrict__ qb, const unsigned short* __restrict__ kb,
    float* __restrict__ part) {
  extern __shared__ unsigned short smem[];
  const int chunk = blockIdx.x;
  const int h = blockIdx.y;
  const int c0 = chunk * CPC;

  const int tid = threadIdx.x, lane = tid & 31, w = tid >> 5;
  const int wm = w >> 2, wn = w & 3;

  const unsigned long long pitch = (unsigned long long)C_DIM * E_DIM;  // 393216

  // base element offset of k-step ks (column c0+cc, d-half dh) for row 0
  auto kbase = [&](int ks) -> size_t {
    return (size_t)(c0 + (ks >> 1)) * E_DIM + h * D_HEAD + (ks & 1) * 32;
  };

  v8f acc[4][2];
#pragma unroll
  for (int r = 0; r < 4; ++r)
#pragma unroll
    for (int s = 0; s < 2; ++s)
#pragma unroll
      for (int j = 0; j < 8; ++j) acc[r][s][j] = 0.0f;

  if (w == 0) {
    tdm_load_2d_bf16(0u,     &qb[kbase(0)], 128, pitch);
    tdm_load_2d_bf16(20480u, &kb[kbase(0)], 128, pitch);
  }
  __builtin_amdgcn_s_wait_tensorcnt(0);
  __syncthreads();

  auto kstep = [&](const unsigned cur, int ks_next, bool prefetch) {
    if (prefetch && w == 0) {
      tdm_load_2d_bf16((cur ^ 1u) * 10240u, &qb[kbase(ks_next)], 128, pitch);
      tdm_load_2d_bf16(20480u + (cur ^ 1u) * 10240u, &kb[kbase(ks_next)], 128, pitch);
    }
    const unsigned short* Qt = smem + cur * 5120u;
    const unsigned short* Kt = smem + 10240u + cur * 5120u;

    v16bf af[4], bfr[2];
#pragma unroll
    for (int r = 0; r < 4; ++r) af[r] = ldsA(&Kt[(wm * 64 + r * 16) * 40], 40, lane);
#pragma unroll
    for (int s = 0; s < 2; ++s) bfr[s] = ldsB(&Qt[(wn * 32 + s * 16) * 40], 40, lane);
#pragma unroll
    for (int r = 0; r < 4; ++r)
#pragma unroll
      for (int s = 0; s < 2; ++s) acc[r][s] = wmma_bf16(af[r], bfr[s], acc[r][s]);

    __builtin_amdgcn_s_wait_tensorcnt(0);
    __syncthreads();
  };

  for (int t = 0; t < 16; ++t) {          // 32 k-steps, unrolled x2
    kstep(0u, 2 * t + 1, true);
    kstep(1u, 2 * t + 2, t < 15);
  }

  const int nloc = lane & 15, hi = lane >> 4;
#pragma unroll
  for (int r = 0; r < 4; ++r) {
    int j0 = wm * 64 + r * 16 + hi * 8;
#pragma unroll
    for (int s = 0; s < 2; ++s) {
      int i = wn * 32 + s * 16 + nloc;
      float* p = &part[(((size_t)chunk * H_DIM + h) * R_DIM + i) * R_DIM + j0];
      *(float4*)p       = make_float4(acc[r][s][0], acc[r][s][1], acc[r][s][2], acc[r][s][3]);
      *(float4*)(p + 4) = make_float4(acc[r][s][4], acc[r][s][5], acc[r][s][6], acc[r][s][7]);
    }
  }
}

// ---------------------------------------------------------------------------
// K3: chunk-reduce + softmax. One wave (32 lanes) per (h,i) row of 128 logits.
// Writes f32 probs to d_out tail and bf16 probs for the context GEMM.
// ---------------------------------------------------------------------------
__global__ __launch_bounds__(256) void softmax_rows(
    const float* __restrict__ part, float* __restrict__ probs_f32,
    unsigned short* __restrict__ probs_bf) {
  const int w = threadIdx.x >> 5, lane = threadIdx.x & 31;
  const int row = blockIdx.x * 8 + w;  // 0..1535
  const int h = row >> 7, i = row & 127;

  float a[4] = {0.f, 0.f, 0.f, 0.f};
  for (int ch = 0; ch < NCHUNK; ++ch) {
    const float* p = &part[(((size_t)ch * H_DIM + h) * R_DIM + i) * R_DIM];
#pragma unroll
    for (int t = 0; t < 4; ++t) a[t] += p[lane + 32 * t];
  }
  float m = fmaxf(fmaxf(a[0], a[1]), fmaxf(a[2], a[3]));
#pragma unroll
  for (int off = 16; off >= 1; off >>= 1) m = fmaxf(m, __shfl_xor(m, off, 32));
  float s = 0.f;
#pragma unroll
  for (int t = 0; t < 4; ++t) { a[t] = __expf(a[t] - m); s += a[t]; }
#pragma unroll
  for (int off = 16; off >= 1; off >>= 1) s += __shfl_xor(s, off, 32);
  float inv = 1.0f / s;
#pragma unroll
  for (int t = 0; t < 4; ++t) {
    float pv = a[t] * inv;
    size_t o = ((size_t)h * R_DIM + i) * R_DIM + lane + 32 * t;
    probs_f32[o] = pv;
    probs_bf[o]  = f2bf(pv);
  }
}

// ---------------------------------------------------------------------------
// K5: context = P @ V per (head, pair of columns). Grid (256, 12).
// D[m=(c_sub,d)][n=i] = sum_j V[(j,c)][h,d] * P[h][i][j];  K=128 j, 4 k-steps.
// P staged once row-major [i][j]; V tile transposed into LDS [(c_sub,d)][j]
// (TDM cannot transpose, so this stays manual). Output features contiguous
// per lane -> 16B bf16 stores.
// ---------------------------------------------------------------------------
__global__ __launch_bounds__(256) void context_gemm(
    const unsigned short* __restrict__ probs_bf,
    const unsigned short* __restrict__ vb, unsigned short* __restrict__ ctx) {
  const int c0 = blockIdx.x * 2;
  const int h = blockIdx.y;

  __shared__ __align__(16) unsigned short Pt[128 * 136];
  __shared__ __align__(16) unsigned short Vt[128 * 40];

  const int tid = threadIdx.x, lane = tid & 31, w = tid >> 5;
  const int wm = w >> 2, wn = w & 3;

  {  // stage P[h] (128x128 bf16) once
    int row = tid >> 1, half = tid & 1;
    const unsigned short* src = &probs_bf[((size_t)h * R_DIM + row) * R_DIM + half * 64];
    unsigned short* dst = &Pt[row * 136 + half * 64];
#pragma unroll
    for (int q8 = 0; q8 < 8; ++q8) *(uint4*)&dst[q8 * 8] = *(const uint4*)&src[q8 * 8];
  }

  v8f acc[4][2];
#pragma unroll
  for (int r = 0; r < 4; ++r)
#pragma unroll
    for (int s = 0; s < 2; ++s)
#pragma unroll
      for (int j = 0; j < 8; ++j) acc[r][s][j] = 0.0f;

  for (int ks = 0; ks < 4; ++ks) {
    const int j0 = ks * 32;
    __syncthreads();  // Pt ready (first iter) / Vt consumed (later iters)
    {                 // stage V transposed: Vt[c_sub*64+d][jj]
      int pair = tid >> 2;
      int jj = pair >> 1, csub = pair & 1;
      int d0 = (tid & 3) * 16;
      const unsigned short* src =
          &vb[((size_t)(j0 + jj) * C_DIM + c0 + csub) * E_DIM + h * D_HEAD + d0];
      union { uint4 u; unsigned short s[8]; } t0, t1;
      t0.u = *(const uint4*)src;
      t1.u = *(const uint4*)(src + 8);
#pragma unroll
      for (int x = 0; x < 8; ++x) Vt[(csub * 64 + d0 + x) * 40 + jj] = t0.s[x];
#pragma unroll
      for (int x = 0; x < 8; ++x) Vt[(csub * 64 + d0 + 8 + x) * 40 + jj] = t1.s[x];
    }
    __syncthreads();
    v16bf af[4], bfr[2];
#pragma unroll
    for (int r = 0; r < 4; ++r) af[r] = ldsA(&Vt[(wm * 64 + r * 16) * 40], 40, lane);
#pragma unroll
    for (int s = 0; s < 2; ++s)
      bfr[s] = ldsB(&Pt[(wn * 32 + s * 16) * 136 + j0], 136, lane);
#pragma unroll
    for (int r = 0; r < 4; ++r)
#pragma unroll
      for (int s = 0; s < 2; ++s) acc[r][s] = wmma_bf16(af[r], bfr[s], acc[r][s]);
  }

  const int nloc = lane & 15, hi = lane >> 4;
#pragma unroll
  for (int r = 0; r < 4; ++r) {
    int m0 = wm * 64 + r * 16 + hi * 8;  // (c_sub,d) index, 8 contiguous d
    int csub = m0 >> 6, d0 = m0 & 63;
#pragma unroll
    for (int s = 0; s < 2; ++s) {
      int i = wn * 32 + s * 16 + nloc;
      float vals[8];
#pragma unroll
      for (int j = 0; j < 8; ++j) vals[j] = acc[r][s][j];
      *(uint4*)&ctx[((size_t)i * C_DIM + c0 + csub) * E_DIM + h * D_HEAD + d0] =
          pack8_bf16(vals);
    }
  }
}

// ---------------------------------------------------------------------------
// Host launcher
// ---------------------------------------------------------------------------
extern "C" void kernel_launch(void* const* d_in, const int* in_sizes, int n_in,
                              void* d_out, int out_size, void* d_ws, size_t ws_size,
                              hipStream_t stream) {
  const float* x  = (const float*)d_in[0];
  const float* Wq = (const float*)d_in[1];
  const float* bq = (const float*)d_in[2];
  const float* Wk = (const float*)d_in[3];
  const float* bk = (const float*)d_in[4];
  const float* Wv = (const float*)d_in[5];
  const float* bv = (const float*)d_in[6];
  const float* Wo = (const float*)d_in[7];
  const float* bo = (const float*)d_in[8];

  float* out_f   = (float*)d_out;                       // [T][768]
  float* probs_f = out_f + (size_t)T_TOK * E_DIM;       // [12][128][128]

  // workspace layout
  size_t off = 0;
  auto carve = [&](size_t bytes) {
    void* p = (char*)d_ws + off;
    off += (bytes + 255) & ~(size_t)255;
    return p;
  };
  const size_t tokE = (size_t)T_TOK * E_DIM;
  unsigned short* xb  = (unsigned short*)carve(tokE * 2);
  unsigned short* qb  = (unsigned short*)carve(tokE * 2);
  unsigned short* kb  = (unsigned short*)carve(tokE * 2);
  unsigned short* vb  = (unsigned short*)carve(tokE * 2);
  unsigned short* ctx = (unsigned short*)carve(tokE * 2);
  unsigned short* wqb = (unsigned short*)carve((size_t)E_DIM * E_DIM * 2);
  unsigned short* wkb = (unsigned short*)carve((size_t)E_DIM * E_DIM * 2);
  unsigned short* wvb = (unsigned short*)carve((size_t)E_DIM * E_DIM * 2);
  unsigned short* wob = (unsigned short*)carve((size_t)E_DIM * E_DIM * 2);
  float* part = (float*)carve((size_t)NCHUNK * H_DIM * R_DIM * R_DIM * 4);
  unsigned short* pbf = (unsigned short*)carve((size_t)H_DIM * R_DIM * R_DIM * 2);

  const float scaling = 5.5242717280199034e-3f;  // D^-0.5 / sqrt(C)
  const unsigned lds_db = 40960;                 // 4 x 10240B double buffers

  // 0: convert inputs to bf16
  {
    int n8 = (int)(tokE / 8);
    cvt_f32_bf16<<<n8 / 256, 256, 0, stream>>>(x, xb, n8);
    int w8 = E_DIM * E_DIM / 8;
    cvt_f32_bf16<<<w8 / 256, 256, 0, stream>>>(Wq, wqb, w8);
    cvt_f32_bf16<<<w8 / 256, 256, 0, stream>>>(Wk, wkb, w8);
    cvt_f32_bf16<<<w8 / 256, 256, 0, stream>>>(Wv, wvb, w8);
    cvt_f32_bf16<<<w8 / 256, 256, 0, stream>>>(Wo, wob, w8);
  }

  // 1: Q/K/V projections (bf16 out, q pre-scaled), TDM-staged
  dim3 gproj(T_TOK / 128, E_DIM / 128);
  gemm_proj<true><<<gproj, 256, lds_db, stream>>>(xb, wqb, bq, qb, scaling);
  gemm_proj<true><<<gproj, 256, lds_db, stream>>>(xb, wkb, bk, kb, 1.0f);
  gemm_proj<true><<<gproj, 256, lds_db, stream>>>(xb, wvb, bv, vb, 1.0f);

  // 2: split-K attention scores, TDM-staged
  scores_partial<<<dim3(NCHUNK, H_DIM), 256, lds_db, stream>>>(qb, kb, part);

  // 3: reduce + softmax (writes probs to d_out tail and bf16 copy)
  softmax_rows<<<(H_DIM * R_DIM) / 8, 256, 0, stream>>>(part, probs_f, pbf);

  // 4: context = P @ V
  context_gemm<<<dim3(C_DIM / 2, H_DIM), 256, 0, stream>>>(pbf, vb, ctx);

  // 5: output projection (f32 out, + bias)
  gemm_proj<false><<<gproj, 256, lds_db, stream>>>(ctx, wob, bo, out_f, 1.0f);
}
